// MMDLoss_18365280157955
// MI455X (gfx1250) — matrix operand order repaired and checked by reading, not compiled, not run on hardware
//
#include <hip/hip_runtime.h>
#include <math.h>

// ---------------- problem constants ----------------
#define N_TOT 8192
#define D     256
#define BHALF 4096
#define TILE  128
#define TBLK  (N_TOT / TILE)   // 64
#define KSTEP 32
#define NKST  (D / KSTEP)      // 8
#define LDS_STRIDE 40          // ushorts per LDS row (padded: conflict-free b128 gathers)

typedef unsigned short u16;
typedef __attribute__((ext_vector_type(16))) __bf16 v16bf;
typedef __attribute__((ext_vector_type(8)))  float  v8f;

union FragB { v16bf v; uint4 q[2]; };
union FragF { v8f   v; float f[8]; };

// ---------------- workspace layout (bytes) ----------------
#define WS_ACC    0        // double: global signed kernel sum
#define WS_NIB    8        // float : -1/(16*bandwidth)
#define WS_SUMSQ  16       // float : sum of row sq-norms
#define WS_COLSUM 64       // float[256]
#define WS_SQ     4096     // float[8192]
#define WS_AHI    40960                       // u16[8192*256]
#define WS_ALO    (40960 + N_TOT * D * 2)     // u16[8192*256]
// total ~8.3 MB

// ---------------- helpers ----------------
__device__ __forceinline__ u16 f2bf_rn(float f) {
    unsigned int u = __float_as_uint(f);
    u += 0x7FFFu + ((u >> 16) & 1u);     // round to nearest even
    return (u16)(u >> 16);
}
__device__ __forceinline__ float bf2f(u16 h) {
    return __uint_as_float(((unsigned int)h) << 16);
}

// ---------------- kernel 0: zero accumulators ----------------
__global__ void mmd_init(double* acc, float* sumsq, float* colsum) {
    int t = threadIdx.x;
    if (t == 0) { *acc = 0.0; *sumsq = 0.0f; }
    if (t < D) colsum[t] = 0.0f;
}

// ---------------- kernel 1: row norms, column sums, bf16 hi/lo split ----------------
// 256 blocks x 256 threads; each wave handles 4 rows.
__global__ void mmd_prep(const float* __restrict__ src, const float* __restrict__ tgt,
                         u16* __restrict__ Ahi, u16* __restrict__ Alo,
                         float* __restrict__ sq, float* __restrict__ colsum,
                         float* __restrict__ sumsq) {
    __shared__ float cs[D];
    __shared__ float bsum;
    int t = threadIdx.x;
    int wave = t >> 5, lane = t & 31;
    if (t < D) cs[t] = 0.0f;
    if (t == 0) bsum = 0.0f;
    __syncthreads();

    float colpart[8];
#pragma unroll
    for (int c = 0; c < 8; ++c) colpart[c] = 0.0f;

    int rowbase = blockIdx.x * 32 + wave * 4;
#pragma unroll
    for (int rr = 0; rr < 4; ++rr) {
        int row = rowbase + rr;
        const float* rp = (row < BHALF) ? (src + (size_t)row * D)
                                        : (tgt + (size_t)(row - BHALF) * D);
        float s = 0.0f;
#pragma unroll
        for (int c = 0; c < 8; ++c) {
            int col = lane + 32 * c;
            float v = rp[col];
            s += v * v;
            colpart[c] += v;
            u16 hi = f2bf_rn(v);
            float lo = v - bf2f(hi);
            Ahi[(size_t)row * D + col] = hi;
            Alo[(size_t)row * D + col] = f2bf_rn(lo);
        }
#pragma unroll
        for (int off = 16; off >= 1; off >>= 1) s += __shfl_xor(s, off, 32);
        if (lane == 0) { sq[row] = s; atomicAdd(&bsum, s); }
    }
#pragma unroll
    for (int c = 0; c < 8; ++c) atomicAdd(&cs[lane + 32 * c], colpart[c]);
    __syncthreads();
    if (t < D) atomicAdd(&colsum[t], cs[t]);
    if (t == 0) atomicAdd(sumsq, bsum);
}

// ---------------- kernel 2: bandwidth (analytic, no GEMM needed) ----------------
__global__ void mmd_bw(const float* __restrict__ colsum, const float* __restrict__ sumsq,
                       float* __restrict__ nib) {
    __shared__ float red[D];
    int t = threadIdx.x;
    float c = colsum[t];
    red[t] = c * c;
    __syncthreads();
    for (int off = 128; off >= 1; off >>= 1) {
        if (t < off) red[t] += red[t + off];
        __syncthreads();
    }
    if (t == 0) {
        double n = (double)N_TOT;
        double bwsum = 2.0 * n * (double)(*sumsq) - 2.0 * (double)red[0];
        double bandwidth = bwsum / (n * n - n) / 4.0;   // / KERNEL_MUL^(5//2)
        *nib = (float)(-1.0 / (16.0 * bandwidth));      // scale of smallest-magnitude exponent
    }
}

// ---------------- kernel 3: tiled bf16x3 WMMA GEMM + kernel-sum epilogue ----------------
// grid (J=64, I=64), triangle J>=I; 512 threads = 16 waves, each wave a 32x32 out tile.
// Double-buffered LDS filled by GLOBAL_LOAD_ASYNC_TO_LDS_B128 (ASYNCcnt DMA path).
__global__ void __launch_bounds__(512)
mmd_gemm(const u16* __restrict__ Ahi, const u16* __restrict__ Alo,
         const float* __restrict__ sq, const float* __restrict__ nibp,
         double* __restrict__ acc) {
    int I = blockIdx.y, J = blockIdx.x;
    if (J < I) return;                    // uniform per block
    int rowI = I * TILE, colJ = J * TILE;

    // [buf][arr][row*stride] ; arr: 0=Ahi, 1=Alo, 2=Bhi, 3=Blo
    __shared__ u16 sBuf[2][4][TILE * LDS_STRIDE];
    __shared__ float sSqI[TILE], sSqJ[TILE];
    __shared__ float sPart[16];

    int t = threadIdx.x;
    int wave = t >> 5, lane = t & 31;
    int l = lane & 15, h16 = lane >> 4;
    int wm = (wave >> 2) * 32, wn = (wave & 3) * 32;

    if (t < TILE)                 sSqI[t]        = sq[rowI + t];
    else if (t < 2 * TILE)        sSqJ[t - TILE] = sq[colJ + (t - TILE)];

    float nib = *nibp;

    FragF accf[2][2];
#pragma unroll
    for (int m = 0; m < 2; ++m)
#pragma unroll
        for (int n = 0; n < 2; ++n)
#pragma unroll
            for (int r = 0; r < 8; ++r) accf[m][n].f[r] = 0.0f;

    // --- async staging setup ---------------------------------------------
    int srow = t >> 2;            // 0..127 : row within tile
    int schu = (t & 3) * 8;       // ushort chunk offset within 32-wide k slice
    const unsigned ARRSZ = (unsigned)(TILE * LDS_STRIDE * 2);   // bytes per array
    const unsigned BUFSZ = 4u * ARRSZ;                          // bytes per buffer
    // low 32 bits of a generic pointer into LDS == wave-relative LDS address
    unsigned ldsBase  = (unsigned)(uintptr_t)(&sBuf[0][0][0]);
    unsigned stageOff = (unsigned)((srow * LDS_STRIDE + schu) * 2);
    unsigned gRowA    = (unsigned)(((rowI + srow) * D + schu) * 2);  // byte offsets
    unsigned gRowB    = (unsigned)(((colJ + srow) * D + schu) * 2);

    auto issue_stage = [&](int kk, int b) {
        unsigned voffA = gRowA + (unsigned)(kk * KSTEP * 2);
        unsigned voffB = gRowB + (unsigned)(kk * KSTEP * 2);
        unsigned l0 = ldsBase + (unsigned)b * BUFSZ + stageOff;
        asm volatile("global_load_async_to_lds_b128 %0, %1, %2"
                     :: "v"(l0),             "v"(voffA), "s"(Ahi) : "memory");
        asm volatile("global_load_async_to_lds_b128 %0, %1, %2"
                     :: "v"(l0 + ARRSZ),     "v"(voffA), "s"(Alo) : "memory");
        asm volatile("global_load_async_to_lds_b128 %0, %1, %2"
                     :: "v"(l0 + 2 * ARRSZ), "v"(voffB), "s"(Ahi) : "memory");
        asm volatile("global_load_async_to_lds_b128 %0, %1, %2"
                     :: "v"(l0 + 3 * ARRSZ), "v"(voffB), "s"(Alo) : "memory");
    };

    // prologue: fill buffer 0
    issue_stage(0, 0);
    asm volatile("s_wait_asynccnt 0x0" ::: "memory");
    __syncthreads();

    for (int kk = 0; kk < NKST; ++kk) {
        int cur = kk & 1;
        if (kk + 1 < NKST) issue_stage(kk + 1, cur ^ 1);   // DMA next stage

        const u16* sAh = &sBuf[cur][0][0];
        const u16* sAl = &sBuf[cur][1][0];
        const u16* sBh = &sBuf[cur][2][0];
        const u16* sBl = &sBuf[cur][3][0];

        // A fragment: lane l holds row (wm+m*16+l), K = {h16*8..+7, 16+h16*8..+7}
        // B fragment: lane l holds col row (wn+n*16+l), K = {h16*8..+7, 16+h16*8..+7}
        FragB fah[2], fal[2], fbh[2], fbl[2];
#pragma unroll
        for (int m = 0; m < 2; ++m) {
            const u16* p = sAh + (wm + m * 16 + l) * LDS_STRIDE;
            fah[m].q[0] = *(const uint4*)(p + h16 * 8);
            fah[m].q[1] = *(const uint4*)(p + 16 + h16 * 8);
            const u16* q = sAl + (wm + m * 16 + l) * LDS_STRIDE;
            fal[m].q[0] = *(const uint4*)(q + h16 * 8);
            fal[m].q[1] = *(const uint4*)(q + 16 + h16 * 8);
        }
#pragma unroll
        for (int n = 0; n < 2; ++n) {
            const u16* p = sBh + (wn + n * 16 + l) * LDS_STRIDE;
            fbh[n].q[0] = *(const uint4*)(p + h16 * 8);
            fbh[n].q[1] = *(const uint4*)(p + 16 + h16 * 8);
            const u16* q = sBl + (wn + n * 16 + l) * LDS_STRIDE;
            fbl[n].q[0] = *(const uint4*)(q + h16 * 8);
            fbl[n].q[1] = *(const uint4*)(q + 16 + h16 * 8);
        }

        // bf16x3 emulated-fp32: hi*hi + hi*lo + lo*hi
#pragma unroll
        for (int m = 0; m < 2; ++m)
#pragma unroll
            for (int n = 0; n < 2; ++n) {
                accf[m][n].v = __builtin_amdgcn_wmma_f32_16x16x32_bf16(
                    false, fah[m].v, false, fbh[n].v, (short)0, accf[m][n].v, false, false);
                accf[m][n].v = __builtin_amdgcn_wmma_f32_16x16x32_bf16(
                    false, fah[m].v, false, fbl[n].v, (short)0, accf[m][n].v, false, false);
                accf[m][n].v = __builtin_amdgcn_wmma_f32_16x16x32_bf16(
                    false, fal[m].v, false, fbh[n].v, (short)0, accf[m][n].v, false, false);
            }

        // wait for next stage's DMA, then sync before swapping buffers
        asm volatile("s_wait_asynccnt 0x0" ::: "memory");
        __syncthreads();
    }

    // epilogue: l2 -> sum of 5 Gaussian kernels via e, e^2, e^4, e^8, e^16
    float tsum = 0.0f;
#pragma unroll
    for (int m = 0; m < 2; ++m)
#pragma unroll
        for (int n = 0; n < 2; ++n) {
            int jl = wn + n * 16 + l;
            float sj = sSqJ[jl];
#pragma unroll
            for (int r = 0; r < 8; ++r) {
                int il = wm + m * 16 + h16 * 8 + r;
                float g  = accf[m][n].f[r];
                float l2 = fmaxf(sSqI[il] + sj - 2.0f * g, 0.0f);
                float e  = __expf(l2 * nib);           // exp(-l2/(16 bw))
                float e2 = e * e, e4 = e2 * e2, e8 = e4 * e4, e16 = e8 * e8;
                tsum += e + e2 + e4 + e8 + e16;
            }
        }
#pragma unroll
    for (int off = 16; off >= 1; off >>= 1) tsum += __shfl_xor(tsum, off, 32);
    if (lane == 0) sPart[wave] = tsum;
    __syncthreads();
    if (t == 0) {
        float total = 0.0f;
#pragma unroll
        for (int w = 0; w < 16; ++w) total += sPart[w];
        float sign = ((I < TBLK / 2) == (J < TBLK / 2)) ? 1.0f : -1.0f;
        float mult = (I == J) ? 1.0f : 2.0f;           // symmetric: count (i,j) and (j,i)
        atomicAdd(acc, (double)(total * sign * mult));
    }
}

// ---------------- kernel 4: final scale ----------------
__global__ void mmd_fin(const double* __restrict__ acc, float* __restrict__ out) {
    out[0] = (float)(*acc / ((double)BHALF * (double)BHALF));
}

// ---------------- launch ----------------
extern "C" void kernel_launch(void* const* d_in, const int* in_sizes, int n_in,
                              void* d_out, int out_size, void* d_ws, size_t ws_size,
                              hipStream_t stream) {
    const float* src = (const float*)d_in[0];
    const float* tgt = (const float*)d_in[1];
    char* wsb = (char*)d_ws;

    double* acc    = (double*)(wsb + WS_ACC);
    float*  nib    = (float*) (wsb + WS_NIB);
    float*  sumsq  = (float*) (wsb + WS_SUMSQ);
    float*  colsum = (float*) (wsb + WS_COLSUM);
    float*  sqv    = (float*) (wsb + WS_SQ);
    u16*    Ahi    = (u16*)   (wsb + WS_AHI);
    u16*    Alo    = (u16*)   (wsb + WS_ALO);

    mmd_init<<<1, 256, 0, stream>>>(acc, sumsq, colsum);
    mmd_prep<<<256, 256, 0, stream>>>(src, tgt, Ahi, Alo, sqv, colsum, sumsq);
    mmd_bw<<<1, 256, 0, stream>>>(colsum, sumsq, nib);
    dim3 grid(TBLK, TBLK);
    mmd_gemm<<<grid, 512, 0, stream>>>(Ahi, Alo, sqv, nib, acc);
    mmd_fin<<<1, 1, 0, stream>>>(acc, (float*)d_out);
}
